// CovidRNN_2104533975456
// MI455X (gfx1250) — matrix operand order repaired and checked by reading, not compiled, not run on hardware
//
#include <hip/hip_runtime.h>
#include <stdint.h>
#include <stddef.h>

// ---------------- problem constants ----------------
#define T_STEPS 8
#define NN      50000
#define EE      800000
#define XD      128
#define HD      256
#define GD      128
#define ZD      128
#define TDIM    8
#define YHD     16
#define PSH     100
// concat widths (K padded to mult of 32)
#define KF      640     // [h(256) | rep(128) | phi(256)]
#define KG      160     // [z(128) | c(8) | yh(16) | pad(8)]

typedef unsigned short u16;
typedef __attribute__((ext_vector_type(16))) __bf16 v16bf;
typedef __attribute__((ext_vector_type(8)))  float  v8f;

__device__ __forceinline__ u16 f2bf(float f) {
    union { float f; uint32_t u; } v; v.f = f;
    uint32_t r = (v.u + 0x7FFFu + ((v.u >> 16) & 1u)) >> 16;
    return (u16)r;
}
__device__ __forceinline__ float sigm(float x) { return 1.0f / (1.0f + expf(-x)); }

// ---------------- weight packing into WMMA B-fragment tiles ----------------
// Tile (kt, ct) of 32(K) x 16(N) bf16 stored as 512 contiguous u16:
// index = lane*16 + e, lane = (k_local/16)*16 + n_local, e = k_local % 16.
__global__ void pack_w_kernel(const float* __restrict__ W, u16* __restrict__ out,
                              int K, int M, int Kt, int Ct4, int transpose) {
    int idx = blockIdx.x * blockDim.x + threadIdx.x;
    int total = Kt * Ct4 * 512;
    if (idx >= total) return;
    int tile   = idx >> 9;
    int within = idx & 511;
    int lane = within >> 4, e = within & 15;
    int kt = tile / Ct4, ct = tile % Ct4;
    int n = ct * 16 + (lane & 15);
    int k = kt * 32 + (lane >> 4) * 16 + e;
    float v = 0.0f;
    if (k < K && n < M)
        v = transpose ? W[(size_t)n * K + k] : W[(size_t)k * M + n];
    out[idx] = f2bf(v);
}

// ---------------- generic bf16 WMMA GEMM ----------------
// 256 thr = 8 waves (2 row-tiles x 4 wave-cols); each wave holds NACC
// accumulators over adjacent 16-col tiles (block = 32 rows x 64*NACC cols).
// A tile staged global->LDS with async copy (ASYNCcnt) already permuted to
// the 16x32 bf16 A-fragment lane layout: for a 4-aligned K group the
// destination (h*16+m)*16+e0.. is 8 contiguous bytes -> one async b64 per
// thread.  Each lane then reads its fragment with a single 32B ds_load pair.
template <int NACC>
__global__ void __launch_bounds__(256)
gemm_bf16_wmma(const u16* __restrict__ A, int lda,
               const u16* __restrict__ Bp,
               const float* __restrict__ bias,
               float* __restrict__ outF, int ldo,
               u16* __restrict__ outB, int ldob, int colOffB,
               int Nrows, int M, int Kpad, int Ct4, int act) {
    __shared__ __align__(32) u16 lds_a[1024];   // two 16-row fragments
    const int tid  = threadIdx.x;
    const int wave = tid >> 5, lane = tid & 31;
    const int wr = wave >> 2, wc = wave & 3;
    const int r0 = blockIdx.x * 32;
    const int ct0 = blockIdx.y * (4 * NACC) + wc * NACC;
    const int cn = lane & 15;

    v8f acc[NACC];
#pragma unroll
    for (int j = 0; j < NACC; ++j) {
        float bv = 0.0f;
        int col = (ct0 + j) * 16 + cn;
        if (bias && col < M) bv = bias[col];
#pragma unroll
        for (int i = 0; i < 8; ++i) acc[j][i] = bv;
    }

    const int Kt = Kpad >> 5;
    const int rl = tid >> 3;            // 0..31 : row within 32-row tile
    const int kq = (tid & 7) << 2;      // 0,4,..,28 : 4 consecutive K
    const int grow = r0 + rl;
    const int frag = rl >> 4, m = rl & 15;
    int h, e0;
    h = kq >> 4; e0 = kq & 15;
    const int lidx = frag * 512 + (h * 16 + m) * 16 + e0;    // u16 units
    const unsigned ldsOff = (unsigned)(lidx * 2);            // LDS byte offset
    const bool rowOk = (grow < Nrows);
    const u16* arow = A + (size_t)grow * lda + kq;

    for (int kt = 0; kt < Kt; ++kt) {
        __syncthreads();                 // prior iteration's fragment reads done
        if (rowOk) {
            unsigned long long ga = (unsigned long long)(arow + kt * 32);
            asm volatile("global_load_async_to_lds_b64 %0, %1, off"
                         :: "v"(ldsOff), "v"(ga) : "memory");
        } else {
            *(uint2*)(&lds_a[lidx]) = make_uint2(0u, 0u);
        }
        asm volatile("s_wait_asynccnt 0x0" ::: "memory");
        __syncthreads();
        v16bf a = *((const v16bf*)(lds_a + wr * 512 + lane * 16));
#pragma unroll
        for (int j = 0; j < NACC; ++j) {
            v16bf b = *((const v16bf*)(Bp + ((size_t)kt * Ct4 + ct0 + j) * 512 + lane * 16));
            acc[j] = __builtin_amdgcn_wmma_f32_16x16x32_bf16(
                         false, a, false, b, (short)0, acc[j], false, false);
        }
    }

#pragma unroll
    for (int j = 0; j < NACC; ++j) {
        int col = (ct0 + j) * 16 + cn;
#pragma unroll
        for (int r = 0; r < 8; ++r) {
            int row = r0 + wr * 16 + r + ((lane >> 4) ? 8 : 0);
            float v = acc[j][r];
            if (act == 1) v = fmaxf(v, 0.0f);
            if (row < Nrows && col < M) {
                if (outF) outF[(size_t)row * ldo + col] = v;
                if (outB) outB[(size_t)row * ldob + colOffB + col] = f2bf(v);
            }
        }
    }
}

// ---------------- elementwise / graph kernels ----------------
__global__ void fill_f32_kernel(float* p, float v, long n) {
    long i = (long)blockIdx.x * blockDim.x + threadIdx.x;
    if (i < n) p[i] = v;
}
__global__ void fill_u16_kernel(u16* p, long n) {
    long i = (long)blockIdx.x * blockDim.x + threadIdx.x;
    if (i < n) p[i] = 0;
}
__global__ void pack_f32_to_bf16(const float* __restrict__ src, u16* __restrict__ dst, long n) {
    long i = (long)blockIdx.x * blockDim.x + threadIdx.x;
    if (i < n) dst[i] = f2bf(src[i]);
}
// pack C[t] (N x 8) and Y_hist[t] (N x 16) into concatG cols 128..151
__global__ void pack_cy_kernel(const float* __restrict__ C, const float* __restrict__ Yh,
                               u16* __restrict__ cg) {
    long i = (long)blockIdx.x * blockDim.x + threadIdx.x;
    if (i >= (long)NN * 24) return;
    int n = (int)(i / 24), j = (int)(i % 24);
    float v = (j < TDIM) ? C[(size_t)n * TDIM + j] : Yh[(size_t)n * YHD + (j - TDIM)];
    cg[(size_t)n * KG + ZD + j] = f2bf(v);
}
__global__ void deg_acc_kernel(const int* __restrict__ dst, const float* __restrict__ ewl,
                               float* __restrict__ deg) {
    int e = blockIdx.x * blockDim.x + threadIdx.x;
    if (e >= EE) return;
    atomicAdd(deg + dst[e], sigm(ewl[e]));
}
// one wave per edge; 32 lanes x float4 = 128-wide payload
__global__ void __launch_bounds__(256)
gcn_scatter_kernel(const int* __restrict__ src, const int* __restrict__ dst,
                   const float* __restrict__ ewl, const float* __restrict__ deg,
                   const float* __restrict__ xw, float* __restrict__ agg) {
    int e = blockIdx.x * 8 + (threadIdx.x >> 5);
    if (e >= EE) return;
    int lane = threadIdx.x & 31;
    int s = src[e], d = dst[e];
    float nrm = rsqrtf(deg[s]) * sigm(ewl[e]) * rsqrtf(deg[d]);
    float4 v = ((const float4*)(xw + (size_t)s * GD))[lane];
    float* ad = agg + (size_t)d * GD + lane * 4;
    atomicAdd(ad + 0, nrm * v.x);
    atomicAdd(ad + 1, nrm * v.y);
    atomicAdd(ad + 2, nrm * v.z);
    atomicAdd(ad + 3, nrm * v.w);
}
// rep = relu(agg + xw/deg + gc_b[t]) -> bf16 into concatF cols 256..383
__global__ void rep_kernel(const float* __restrict__ agg, const float* __restrict__ xw,
                           const float* __restrict__ deg, const float* __restrict__ gcb,
                           u16* __restrict__ cf) {
    long i = (long)blockIdx.x * blockDim.x + threadIdx.x;
    if (i >= (long)NN * GD) return;
    int n = (int)(i / GD), d = (int)(i % GD);
    float v = agg[i] + xw[i] / deg[n] + gcb[d];
    cf[(size_t)n * KF + HD + d] = f2bf(fmaxf(v, 0.0f));
}
// GRU gates: h_new = (1-u)*tanh(i_n + r*h_n) + u*h ; update h f32 + bf16 in concatF
__global__ void gru_gate_kernel(const float* __restrict__ gi, const float* __restrict__ gh,
                                float* __restrict__ h, u16* __restrict__ cf) {
    long i = (long)blockIdx.x * blockDim.x + threadIdx.x;
    if (i >= (long)NN * HD) return;
    int n = (int)(i / HD), d = (int)(i % HD);
    size_t b = (size_t)n * (3 * HD);
    float r  = sigm(gi[b + d]        + gh[b + d]);
    float u  = sigm(gi[b + HD + d]   + gh[b + HD + d]);
    float nn = tanhf(gi[b + 2*HD + d] + r * gh[b + 2*HD + d]);
    float hn = (1.0f - u) * nn + u * h[i];
    h[i] = hn;
    cf[(size_t)n * KF + d] = f2bf(hn);
}
// y = tbuf[n,:] . w + b  (128-wide dot per node)
__global__ void head_dot_kernel(const float* __restrict__ tbuf, const float* __restrict__ w,
                                const float* __restrict__ b, float* __restrict__ out) {
    int n = blockIdx.x * blockDim.x + threadIdx.x;
    if (n >= NN) return;
    const float4* tp = (const float4*)(tbuf + (size_t)n * ZD);
    const float4* wp = (const float4*)w;
    float s = 0.0f;
#pragma unroll
    for (int i = 0; i < ZD / 4; ++i) {
        float4 a = tp[i], c = wp[i];
        s += a.x * c.x + a.y * c.y + a.z * c.z + a.w * c.w;
    }
    out[n] = s + b[0];
}
// propensity: s=sigmoid(pre*bn_scale+bn_b); softmax(s@ps2_W + ps2_b)
__global__ void ps_kernel(const float* __restrict__ spre, const float* __restrict__ bng,
                          const float* __restrict__ bnb, const float* __restrict__ w2,
                          const float* __restrict__ b2, float* __restrict__ out) {
    int n = blockIdx.x * blockDim.x + threadIdx.x;
    if (n >= NN) return;
    const float inv = rsqrtf(1.0f + 1e-5f);
    float l0 = b2[0], l1 = b2[1];
    const float* sp = spre + (size_t)n * PSH;
    for (int j = 0; j < PSH; ++j) {
        float s = sigm(sp[j] * (bng[j] * inv) + bnb[j]);
        l0 += s * w2[j * 2 + 0];
        l1 += s * w2[j * 2 + 1];
    }
    float m = fmaxf(l0, l1);
    float e0 = expf(l0 - m), e1 = expf(l1 - m);
    float den = e0 + e1;
    out[(size_t)n * 2 + 0] = e0 / den;
    out[(size_t)n * 2 + 1] = e1 / den;
}
__global__ void copy_f32_kernel(const float* __restrict__ src, float* __restrict__ dst, long n) {
    long i = (long)blockIdx.x * blockDim.x + threadIdx.x;
    if (i < n) dst[i] = src[i];
}

// ---------------- host-side orchestration ----------------
static inline int cdiv(long a, long b) { return (int)((a + b - 1) / b); }

static void run_gemm(const u16* A, int lda, const u16* Bp, const float* bias,
                     float* outF, int ldo, u16* outB, int ldob, int colOff,
                     int M, int Kpad, int Ct4, int act, int nacc, hipStream_t s) {
    dim3 grid(cdiv(NN, 32), Ct4 / (4 * nacc));
    if (nacc == 4)
        gemm_bf16_wmma<4><<<grid, 256, 0, s>>>(A, lda, Bp, bias, outF, ldo,
                                               outB, ldob, colOff, NN, M, Kpad, Ct4, act);
    else
        gemm_bf16_wmma<2><<<grid, 256, 0, s>>>(A, lda, Bp, bias, outF, ldo,
                                               outB, ldob, colOff, NN, M, Kpad, Ct4, act);
}

extern "C" void kernel_launch(void* const* d_in, const int* in_sizes, int n_in,
                              void* d_out, int out_size, void* d_ws, size_t ws_size,
                              hipStream_t stream) {
    // ---- inputs (setup_inputs order) ----
    const float* X     = (const float*)d_in[0];
    const float* C     = (const float*)d_in[1];
    const float* Yh    = (const float*)d_in[2];
    const int*   EI    = (const int*)d_in[3];
    const float* phiW  = (const float*)d_in[4];
    const float* phib  = (const float*)d_in[5];
    const float* gcW   = (const float*)d_in[6];
    const float* gcb   = (const float*)d_in[7];
    const float* ewl   = (const float*)d_in[8];
    const float* fuseW = (const float*)d_in[9];
    const float* fuseb = (const float*)d_in[10];
    const float* o00W  = (const float*)d_in[11];
    const float* o00b  = (const float*)d_in[12];
    const float* o10W  = (const float*)d_in[13];
    const float* o10b  = (const float*)d_in[14];
    const float* o01W  = (const float*)d_in[15];
    const float* o01b  = (const float*)d_in[16];
    const float* o11W  = (const float*)d_in[17];
    const float* o11b  = (const float*)d_in[18];
    const float* ps1W  = (const float*)d_in[19];
    const float* ps1b  = (const float*)d_in[20];
    const float* bng   = (const float*)d_in[21];
    const float* bnb   = (const float*)d_in[22];
    const float* ps2W  = (const float*)d_in[23];
    const float* ps2b  = (const float*)d_in[24];
    const float* Wih   = (const float*)d_in[25];
    const float* Whh   = (const float*)d_in[26];
    const float* bih   = (const float*)d_in[27];
    const float* bhh   = (const float*)d_in[28];

    // ---- outputs: (y1, y0, z, ps, h_fin) flat ----
    float* out = (float*)d_out;
    const long off_y1 = 0;
    const long off_y0 = (long)T_STEPS * NN;
    const long off_z  = off_y0 + (long)T_STEPS * NN;
    const long off_ps = off_z + (long)T_STEPS * NN * ZD;
    const long off_h  = off_ps + (long)T_STEPS * NN * 2;

    // ---- workspace carve-up (256B aligned) ----
    char* base = (char*)d_ws;
    size_t off = 0;
    auto carve = [&](size_t bytes) -> char* {
        char* p = base + off;
        off = (off + bytes + 255) & ~(size_t)255;
        return p;
    };
    u16* wpPhi  = (u16*)carve(4  * 16 * 512 * 2);                 // K128 -> M256
    u16* wpGc   = (u16*)carve((size_t)T_STEPS * 8 * 8 * 512 * 2); // K256 -> M128, per t
    u16* wpFuse = (u16*)carve(20 * 8  * 512 * 2);                 // K640 -> M128
    u16* wpWih  = (u16*)carve(5  * 48 * 512 * 2);                 // K152p160 -> M768 (T)
    u16* wpWhh  = (u16*)carve(8  * 48 * 512 * 2);                 // K256 -> M768 (T)
    u16* wpO00  = (u16*)carve(4  * 8  * 512 * 2);
    u16* wpO10  = (u16*)carve(4  * 8  * 512 * 2);
    u16* wpPs1  = (u16*)carve(4  * 8  * 512 * 2);                 // K128 -> M100
    u16*   Xbf  = (u16*)carve((size_t)NN * XD * 2);
    u16*   cF   = (u16*)carve((size_t)NN * KF * 2);               // [h|rep|phi] bf16
    u16*   cG   = (u16*)carve((size_t)NN * KG * 2);               // [z|c|yh|0] bf16
    float* xw   = (float*)carve((size_t)NN * GD * 4);
    float* deg  = (float*)carve((size_t)NN * 4);
    float* agg  = (float*)carve((size_t)NN * GD * 4);
    float* gi   = (float*)carve((size_t)NN * 3 * HD * 4);
    float* gh   = (float*)carve((size_t)NN * 3 * HD * 4);
    float* hbuf = (float*)carve((size_t)NN * HD * 4);
    float* tbuf = (float*)carve((size_t)NN * ZD * 4);
    float* spre = (float*)carve((size_t)NN * PSH * 4);
    (void)ws_size; (void)n_in; (void)in_sizes; (void)out_size;

    const int BT = 256;
    // ---- pack all weights (once per launch) ----
    pack_w_kernel<<<cdiv(4*16*512, BT),  BT, 0, stream>>>(phiW,  wpPhi,  XD,  HD,  4, 16, 0);
    for (int t = 0; t < T_STEPS; ++t)
        pack_w_kernel<<<cdiv(8*8*512, BT), BT, 0, stream>>>(gcW + (size_t)t*HD*GD,
                                                            wpGc + (size_t)t*8*8*512,
                                                            HD, GD, 8, 8, 0);
    pack_w_kernel<<<cdiv(20*8*512, BT),  BT, 0, stream>>>(fuseW, wpFuse, KF,  ZD,  20, 8, 0);
    pack_w_kernel<<<cdiv(5*48*512, BT),  BT, 0, stream>>>(Wih,   wpWih,  ZD+TDIM+YHD, 3*HD, 5, 48, 1);
    pack_w_kernel<<<cdiv(8*48*512, BT),  BT, 0, stream>>>(Whh,   wpWhh,  HD,  3*HD, 8, 48, 1);
    pack_w_kernel<<<cdiv(4*8*512, BT),   BT, 0, stream>>>(o00W,  wpO00,  ZD,  ZD,  4, 8, 0);
    pack_w_kernel<<<cdiv(4*8*512, BT),   BT, 0, stream>>>(o10W,  wpO10,  ZD,  ZD,  4, 8, 0);
    pack_w_kernel<<<cdiv(4*8*512, BT),   BT, 0, stream>>>(ps1W,  wpPs1,  ZD,  PSH, 4, 8, 0);

    // ---- init state ----
    fill_f32_kernel<<<cdiv((long)NN*HD, BT), BT, 0, stream>>>(hbuf, 0.0f, (long)NN*HD);
    fill_u16_kernel<<<cdiv((long)NN*KF, BT), BT, 0, stream>>>(cF, (long)NN*KF);
    fill_u16_kernel<<<cdiv((long)NN*KG, BT), BT, 0, stream>>>(cG, (long)NN*KG);

    for (int t = 0; t < T_STEPS; ++t) {
        const float* Xt  = X  + (size_t)t * NN * XD;
        const float* Ct  = C  + (size_t)t * NN * TDIM;
        const float* Yt  = Yh + (size_t)t * NN * YHD;
        const int*   src = EI + (size_t)t * 2 * EE;
        const int*   dst = src + EE;
        const float* ewt = ewl + (size_t)t * EE;
        const float* gbt = gcb + (size_t)t * GD;

        // phi = relu(X@phiW + b) -> bf16 into concatF cols 384..639
        pack_f32_to_bf16<<<cdiv((long)NN*XD, BT), BT, 0, stream>>>(Xt, Xbf, (long)NN*XD);
        run_gemm(Xbf, XD, wpPhi, phib, nullptr, 0, cF, KF, HD + GD,
                 HD, XD, 16, 1, 4, stream);

        // xw = phi @ gc_W[t]   (A = concatF phi slice, lda=KF)
        run_gemm(cF + HD + GD, KF, wpGc + (size_t)t*8*8*512, nullptr,
                 xw, GD, nullptr, 0, 0, GD, HD, 8, 0, 2, stream);

        // degrees + scatter aggregation
        fill_f32_kernel<<<cdiv((long)NN, BT), BT, 0, stream>>>(deg, 1.0f, (long)NN);
        deg_acc_kernel<<<cdiv(EE, BT), BT, 0, stream>>>(dst, ewt, deg);
        fill_f32_kernel<<<cdiv((long)NN*GD, BT), BT, 0, stream>>>(agg, 0.0f, (long)NN*GD);
        gcn_scatter_kernel<<<cdiv(EE, 8), 256, 0, stream>>>(src, dst, ewt, deg, xw, agg);
        rep_kernel<<<cdiv((long)NN*GD, BT), BT, 0, stream>>>(agg, xw, deg, gbt, cF);

        // z = relu(concatF @ fuseW + b) -> f32 to d_out, bf16 to concatG cols 0..127
        run_gemm(cF, KF, wpFuse, fuseb,
                 out + off_z + (size_t)t * NN * ZD, ZD, cG, KG, 0,
                 ZD, KF, 8, 1, 2, stream);

        // GRU input tail + gi/gh GEMMs (gh reads h_{t-1} bf16 = concatF cols 0..255)
        pack_cy_kernel<<<cdiv((long)NN*24, BT), BT, 0, stream>>>(Ct, Yt, cG);
        run_gemm(cG, KG, wpWih, bih, gi, 3*HD, nullptr, 0, 0, 3*HD, KG, 48, 0, 4, stream);
        run_gemm(cF, KF, wpWhh, bhh, gh, 3*HD, nullptr, 0, 0, 3*HD, HD, 48, 0, 4, stream);
        gru_gate_kernel<<<cdiv((long)NN*HD, BT), BT, 0, stream>>>(gi, gh, hbuf, cF);

        // outcome heads (A = z bf16 in concatG, lda=KG)
        run_gemm(cG, KG, wpO00, o00b, tbuf, ZD, nullptr, 0, 0, ZD, XD, 8, 1, 2, stream);
        head_dot_kernel<<<cdiv(NN, BT), BT, 0, stream>>>(tbuf, o01W, o01b, out + off_y0 + (size_t)t*NN);
        run_gemm(cG, KG, wpO10, o10b, tbuf, ZD, nullptr, 0, 0, ZD, XD, 8, 1, 2, stream);
        head_dot_kernel<<<cdiv(NN, BT), BT, 0, stream>>>(tbuf, o11W, o11b, out + off_y1 + (size_t)t*NN);

        // propensity head
        run_gemm(cG, KG, wpPs1, ps1b, spre, PSH, nullptr, 0, 0, PSH, XD, 8, 0, 2, stream);
        ps_kernel<<<cdiv(NN, BT), BT, 0, stream>>>(spre, bng, bnb, ps2W, ps2b,
                                                   out + off_ps + (size_t)t*NN*2);
    }

    copy_f32_kernel<<<cdiv((long)NN*HD, BT), BT, 0, stream>>>(hbuf, out + off_h, (long)NN*HD);
}
